// hiifv3_35966056137253
// MI455X (gfx1250) — compile-verified, matrix operand order checked by reading, not run on hardware
//
#include <hip/hip_runtime.h>
#include <hip/hip_bf16.h>
#include <math.h>

typedef __attribute__((ext_vector_type(16))) _Float16 v16h;
typedef __attribute__((ext_vector_type(8)))  float    v8f;
typedef __attribute__((ext_vector_type(4)))  int      i32x4;
typedef __attribute__((address_space(1))) i32x4 as1_i32x4;
typedef __attribute__((address_space(3))) i32x4 as3_i32x4;

#define HW_TOK 65536
#define NPIX   4096

#if __has_builtin(__builtin_amdgcn_global_load_async_to_lds_b128)
#define USE_ASYNC_LDS 1
#else
#define USE_ASYNC_LDS 0
#endif

enum { EPI_F16 = 0, EPI_GELU16 = 1, EPI_F32 = 2 };

__device__ __forceinline__ float geluf(float x) {
  return 0.5f * x * (1.0f + erff(x * 0.70710678118654752f));
}

// ---------------------------------------------------------------- convs ----
__global__ void conv3x3_k(const float* __restrict__ in, const float* __restrict__ w,
                          const float* __restrict__ bias, float* __restrict__ out,
                          int Cin, int Cout) {
  int i = blockIdx.x * 256 + threadIdx.x;
  if (i >= Cout * NPIX) return;
  int co = i >> 12, pix = i & 4095;
  int py = pix >> 6, px = pix & 63;
  float s = bias[co];
  for (int ci = 0; ci < Cin; ++ci) {
    const float* ip = in + (size_t)ci * NPIX;
    const float* wp = w + ((size_t)co * Cin + ci) * 9;
#pragma unroll
    for (int ky = 0; ky < 3; ++ky) {
      int y = py + ky - 1;
      if (y < 0 || y > 63) continue;
#pragma unroll
      for (int kx = 0; kx < 3; ++kx) {
        int x = px + kx - 1;
        if (x < 0 || x > 63) continue;
        s += ip[y * 64 + x] * wp[ky * 3 + kx];
      }
    }
  }
  out[i] = s;
}

// ------------------------------------------------- nearest sample + areas ----
__global__ void samp_k(const float* __restrict__ coord, int* __restrict__ idx4,
                       float* __restrict__ w4, float* __restrict__ lc) {
  int t = blockIdx.x * 256 + threadIdx.x;
  if (t >= HW_TOK) return;
  float cy = coord[t * 2 + 0], cx = coord[t * 2 + 1];
  const float r = 1.f / 64.f, eps = 1e-6f;
  float areas[4]; int idxs[4];
  float lcy = 0.f, lcx = 0.f;
#pragma unroll
  for (int j = 0; j < 4; ++j) {
    float vx = (j < 2) ? -1.f : 1.f;
    float vy = (j & 1) ? 1.f : -1.f;
    float sy = fminf(fmaxf(cy + vx * r + eps, -1.f + 1e-6f), 1.f - 1e-6f);
    float sx = fminf(fmaxf(cx + vy * r + eps, -1.f + 1e-6f), 1.f - 1e-6f);
    int iy = (int)floorf(((sy + 1.f) * 64.f - 1.f) * 0.5f + 0.5f);
    int ix = (int)floorf(((sx + 1.f) * 64.f - 1.f) * 0.5f + 0.5f);
    iy = min(max(iy, 0), 63); ix = min(max(ix, 0), 63);
    idxs[j] = iy * 64 + ix;
    float oy = -1.f + r + 2.f * r * (float)iy;
    float ox = -1.f + r + 2.f * r * (float)ix;
    float ry_ = (cy - oy) * 32.f, rx_ = (cx - ox) * 32.f;
    areas[j] = fabsf(ry_ * rx_) + 1e-9f;
    if (j == 0) {
      lcy = (ry_ > 0.f) ? ry_ : (r - ry_);
      lcx = (rx_ > 0.f) ? rx_ : (r - rx_);
    }
  }
  float tot = areas[0] + areas[1] + areas[2] + areas[3];
#pragma unroll
  for (int j = 0; j < 4; ++j) {
    idx4[t * 4 + j] = idxs[j];
    w4[t * 4 + j] = areas[3 - j] / tot;
  }
  lc[t * 2 + 0] = lcy;
  lc[t * 2 + 1] = lcx;
}

// ---------------------------------------- layer0 fused gather+LN -> f16 A ----
__global__ __launch_bounds__(256) void ln_big(
    const float* __restrict__ feat, const int* __restrict__ idx4,
    const float* __restrict__ w4, const float* __restrict__ lc,
    const float* __restrict__ cell, const float* __restrict__ g,
    const float* __restrict__ b, _Float16* __restrict__ A) {
  const int lane = threadIdx.x & 31;
  const int t = blockIdx.x * 8 + (threadIdx.x >> 5);
  const int i0 = idx4[t * 4 + 0], i1 = idx4[t * 4 + 1],
            i2 = idx4[t * 4 + 2], i3 = idx4[t * 4 + 3];
  const float w0 = w4[t * 4 + 0], w1 = w4[t * 4 + 1],
              w2 = w4[t * 4 + 2], w3 = w4[t * 4 + 3];
  float vals[32];
  float S = 0.f, SS = 0.f;
#pragma unroll
  for (int u = 0; u < 32; ++u) {
    int e = lane + u * 32;
    int j = e >> 8, c = e & 255;
    int idx = (j == 0) ? i0 : (j == 1) ? i1 : (j == 2) ? i2 : i3;
    float w = (j == 0) ? w0 : (j == 1) ? w1 : (j == 2) ? w2 : w3;
    float v = feat[(size_t)c * NPIX + idx] * w;
    vals[u] = v; S += v; SS += v * v;
  }
  float ex = 0.f;
  if (lane < 2) ex = cell[lane] * 64.f;
  else if (lane < 4) {
    float l = lc[t * 2 + (lane - 2)];
    float cl = fminf(fmaxf(l - 1e-9f, 0.f), 1.f);
    ex = fmodf(floorf(cl * 2.f), 2.f);
  }
  if (lane < 4) { S += ex; SS += ex * ex; }
#pragma unroll
  for (int o = 16; o > 0; o >>= 1) { S += __shfl_xor(S, o); SS += __shfl_xor(SS, o); }
  const float inv = 1.f / 1028.f;
  float mean = S * inv, var = SS * inv - mean * mean;
  float rs = rsqrtf(var + 1e-5f);
  _Float16* row = A + (size_t)t * 1056;
#pragma unroll
  for (int u = 0; u < 32; ++u) {
    int e = lane + u * 32;
    row[e] = (_Float16)((vals[u] - mean) * rs * g[e] + b[e]);
  }
  if (lane < 4) {
    int e = 1024 + lane;
    row[e] = (_Float16)((ex - mean) * rs * g[e] + b[e]);
  }
  if (lane < 28) row[1028 + lane] = (_Float16)0.f;
}

// ------------------------------------- layers 1..5 LN(258) -> f16 A[288] ----
__global__ __launch_bounds__(256) void ln_small(
    const float* __restrict__ X, const float* __restrict__ lc,
    const float* __restrict__ g, const float* __restrict__ b,
    _Float16* __restrict__ A, int n) {
  const int lane = threadIdx.x & 31;
  const int t = blockIdx.x * 8 + (threadIdx.x >> 5);
  float vals[8];
  float S = 0.f, SS = 0.f;
#pragma unroll
  for (int u = 0; u < 8; ++u) {
    float v = X[(size_t)t * 256 + lane + u * 32];
    vals[u] = v; S += v; SS += v * v;
  }
  float ex = 0.f;
  if (lane < 2) {
    float l = lc[t * 2 + lane];
    float cl = fminf(fmaxf(l - 1e-9f, 0.f), 1.f);
    ex = fmodf(floorf(cl * (float)(1 << (n + 1))), 2.f);
    S += ex; SS += ex * ex;
  }
#pragma unroll
  for (int o = 16; o > 0; o >>= 1) { S += __shfl_xor(S, o); SS += __shfl_xor(SS, o); }
  const float inv = 1.f / 258.f;
  float mean = S * inv, var = SS * inv - mean * mean;
  float rs = rsqrtf(var + 1e-5f);
  _Float16* row = A + (size_t)t * 288;
#pragma unroll
  for (int u = 0; u < 8; ++u) {
    int e = lane + u * 32;
    row[e] = (_Float16)((vals[u] - mean) * rs * g[e] + b[e]);
  }
  if (lane < 2) {
    int e = 256 + lane;
    row[e] = (_Float16)((ex - mean) * rs * g[e] + b[e]);
  }
  if (lane < 30) row[258 + lane] = (_Float16)0.f;
}

// ------------------------------------------------------------ f32 -> f16 ----
__global__ void cvt16_k(const float* __restrict__ X, _Float16* __restrict__ A, int n) {
  int i = blockIdx.x * 256 + threadIdx.x;
  if (i < n) A[i] = (_Float16)X[i];
}

// ------------------- weight f32 [K,N] -> f16 in B-fragment-native layout ----
__global__ void wcvt_k(const float* __restrict__ W, _Float16* __restrict__ O,
                       int K, int N, int Kp, int Np) {
  int i = blockIdx.x * 256 + threadIdx.x;
  if (i >= Kp * Np) return;
  int e = i & 15;
  int lane = (i >> 4) & 31;
  int blob = i >> 9;
  int kb = Kp >> 5;
  int k32 = blob % kb;
  int n16 = blob / kb;
  int col = n16 * 16 + (lane & 15);
  int kidx = (k32 << 5) + ((e >> 3) << 4) + ((lane >> 4) << 3) + (e & 7);
  float v = (col < N && kidx < K) ? W[(size_t)kidx * N + col] : 0.f;
  O[i] = (_Float16)v;
}

// ------------------------------------------------------------- WMMA GEMM ----
// C[65536 x Np] = A[65536 x Kp](f16) @ Wf(fragment layout) + bias, epilogue.
// Block 256 thr = 8 waves, tile 128x64, wave tile 32x32 (2x2 wmma accum).
// A tile staged via GLOBAL_LOAD_ASYNC_TO_LDS (ASYNCcnt) when available.
__global__ __launch_bounds__(256) void gemm_wmma(
    const _Float16* __restrict__ A, const _Float16* __restrict__ Wf,
    const float* __restrict__ bias, void* __restrict__ C,
    const float* __restrict__ Res, int Kp, int Np, int Nreal, int epi) {
  __shared__ _Float16 lA[128 * 32];
  const int tid = threadIdx.x;
  const int wave = tid >> 5, lane = tid & 31;
  const int bm = blockIdx.x * 128;
  const int bn = blockIdx.y * 64;
  const int wm = (wave & 3) * 32;
  const int wn = (wave >> 2) * 32;
  const int lrow = lane & 15;
  const int khi = (lane >> 4) << 3;

  v8f acc00 = {}, acc01 = {}, acc10 = {}, acc11 = {};
  const int kb = Kp >> 5;

  for (int ks = 0; ks < kb; ++ks) {
    const int kk = ks << 5;
    __syncthreads();
#pragma unroll
    for (int it = 0; it < 2; ++it) {
      int i = tid + it * 256;
      int r = i >> 2, c8 = (i & 3) << 3;
      const _Float16* gsrc = &A[(size_t)(bm + r) * Kp + kk + c8];
#if USE_ASYNC_LDS
      __builtin_amdgcn_global_load_async_to_lds_b128(
          (as1_i32x4*)(void*)gsrc, (as3_i32x4*)(void*)&lA[r * 32 + c8], 0, 0);
#else
      *(uint4*)&lA[r * 32 + c8] = *(const uint4*)gsrc;
#endif
    }
    // prefetch next K tile of A into cache while this one is consumed
    if (ks + 1 < kb)
      __builtin_prefetch(&A[(size_t)(bm + (tid >> 1)) * Kp + kk + 32 + (tid & 1) * 16], 0, 0);
#if USE_ASYNC_LDS
#if __has_builtin(__builtin_amdgcn_s_wait_asynccnt)
    __builtin_amdgcn_s_wait_asynccnt(0);
#else
    asm volatile("s_wait_asynccnt 0x0" ::: "memory");
#endif
#endif
    __syncthreads();

    union U { uint4 u[2]; v16h h; } fa0, fa1, fb0, fb1;
    {
      const _Float16* p0 = &lA[(wm + lrow) * 32 + khi];
      fa0.u[0] = *(const uint4*)p0;       fa0.u[1] = *(const uint4*)(p0 + 16);
      const _Float16* p1 = &lA[(wm + 16 + lrow) * 32 + khi];
      fa1.u[0] = *(const uint4*)p1;       fa1.u[1] = *(const uint4*)(p1 + 16);
    }
    {
      int n16a = (bn + wn) >> 4;
      const _Float16* q0 = Wf + (((size_t)(n16a * kb + ks)) << 9) + lane * 16;
      fb0.u[0] = *(const uint4*)q0;       fb0.u[1] = *(const uint4*)(q0 + 8);
      const _Float16* q1 = Wf + (((size_t)((n16a + 1) * kb + ks)) << 9) + lane * 16;
      fb1.u[0] = *(const uint4*)q1;       fb1.u[1] = *(const uint4*)(q1 + 8);
    }
    acc00 = __builtin_amdgcn_wmma_f32_16x16x32_f16(false, fa0.h, false, fb0.h, (short)0, acc00, false, false);
    acc01 = __builtin_amdgcn_wmma_f32_16x16x32_f16(false, fa0.h, false, fb1.h, (short)0, acc01, false, false);
    acc10 = __builtin_amdgcn_wmma_f32_16x16x32_f16(false, fa1.h, false, fb0.h, (short)0, acc10, false, false);
    acc11 = __builtin_amdgcn_wmma_f32_16x16x32_f16(false, fa1.h, false, fb1.h, (short)0, acc11, false, false);
  }

  const int row_base = bm + wm + ((lane >> 4) << 3);
  const int col0 = bn + wn + lrow;
  const float bia0 = bias ? ((col0 < Nreal) ? bias[col0] : 0.f) : 0.f;
  const float bia1 = bias ? ((col0 + 16 < Nreal) ? bias[col0 + 16] : 0.f) : 0.f;

  auto store = [&](int row, int col, float v) {
    size_t o = (size_t)row * Np + col;
    if (epi == EPI_GELU16) ((_Float16*)C)[o] = (_Float16)geluf(v);
    else if (epi == EPI_F16) ((_Float16*)C)[o] = (_Float16)v;
    else { if (Res) v += Res[o]; ((float*)C)[o] = v; }
  };
#pragma unroll
  for (int r = 0; r < 8; ++r) {
    store(row_base + r,      col0,      acc00[r] + bia0);
    store(row_base + r,      col0 + 16, acc01[r] + bia1);
    store(row_base + 16 + r, col0,      acc10[r] + bia0);
    store(row_base + 16 + r, col0 + 16, acc11[r] + bia1);
  }
}

// ------------------------------------------------ linear-attention pieces ----
__global__ void kv_zero_k(float* kv) { kv[blockIdx.x * 256 + threadIdx.x] = 0.f; }

__global__ __launch_bounds__(256) void kv_reduce(
    const _Float16* __restrict__ QKV, const float* __restrict__ kg,
    const float* __restrict__ kb, const float* __restrict__ vg,
    const float* __restrict__ vb, float* __restrict__ kvout, int tpb) {
  const int h = threadIdx.x >> 4;
  const int d = threadIdx.x & 15;
  float acc[16];
#pragma unroll
  for (int e = 0; e < 16; ++e) acc[e] = 0.f;
  float vge[16], vbe[16];
#pragma unroll
  for (int e = 0; e < 16; ++e) { vge[e] = vg[e]; vbe[e] = vb[e]; }
  const float kgd = kg[d], kbd = kb[d];
  for (int tt = 0; tt < tpb; ++tt) {
    int t = blockIdx.x * tpb + tt;
    const _Float16* base = QKV + (size_t)t * 768 + h * 48;
    float ks = 0, ks2 = 0, vs = 0, vs2 = 0, kdr = 0;
    float vr[16];
#pragma unroll
    for (int i = 0; i < 16; ++i) {
      float kvl = (float)base[16 + i];
      float vvl = (float)base[32 + i];
      ks += kvl; ks2 += kvl * kvl;
      vs += vvl; vs2 += vvl * vvl;
      if (i == d) kdr = kvl;
      vr[i] = vvl;
    }
    float kmean = ks * 0.0625f, kvar = ks2 * 0.0625f - kmean * kmean;
    float vmean = vs * 0.0625f, vvar = vs2 * 0.0625f - vmean * vmean;
    float kn = (kdr - kmean) * rsqrtf(kvar + 1e-5f) * kgd + kbd;
    float vrs = rsqrtf(vvar + 1e-5f);
#pragma unroll
    for (int e = 0; e < 16; ++e)
      acc[e] += kn * ((vr[e] - vmean) * vrs * vge[e] + vbe[e]);
  }
#pragma unroll
  for (int e = 0; e < 16; ++e)
    atomicAdd(&kvout[(h * 16 + d) * 16 + e], acc[e]);
}

__global__ void apply_o_k(const _Float16* __restrict__ QKV, const float* __restrict__ kv,
                          const float* __restrict__ X, float* __restrict__ Y,
                          _Float16* __restrict__ A) {
  int i = blockIdx.x * 256 + threadIdx.x;
  int t = i >> 8, c = i & 255, h = c >> 4, e = c & 15;
  const _Float16* q = QKV + (size_t)t * 768 + h * 48;
  const float* kvh = kv + (size_t)(h * 16) * 16 + e;
  float s = 0.f;
#pragma unroll
  for (int d = 0; d < 16; ++d) s += (float)q[d] * kvh[d * 16];
  s = s * (1.f / 65536.f) + X[i];
  Y[i] = s;
  A[i] = (_Float16)s;
}

// -------------------------------------------- final: result + bilinear up ----
__global__ void finalk_k(const float* __restrict__ Y, const float* __restrict__ inp,
                         const float* __restrict__ coord, float* __restrict__ out) {
  int t = blockIdx.x * 256 + threadIdx.x;
  float cy = coord[t * 2 + 0], cx = coord[t * 2 + 1];
  float yf = ((cy + 1.f) * 64.f - 1.f) * 0.5f;
  float xf = ((cx + 1.f) * 64.f - 1.f) * 0.5f;
  float y0 = floorf(yf), x0 = floorf(xf);
  float wy = yf - y0, wx = xf - x0;
  int y0i = min(max((int)y0, 0), 63), y1i = min(max((int)y0 + 1, 0), 63);
  int x0i = min(max((int)x0, 0), 63), x1i = min(max((int)x0 + 1, 0), 63);
#pragma unroll
  for (int c = 0; c < 3; ++c) {
    const float* im = inp + (size_t)c * NPIX;
    float v = im[y0i * 64 + x0i] * (1.f - wy) * (1.f - wx)
            + im[y0i * 64 + x1i] * (1.f - wy) * wx
            + im[y1i * 64 + x0i] * wy * (1.f - wx)
            + im[y1i * 64 + x1i] * wy * wx;
    out[(size_t)c * HW_TOK + t] = Y[(size_t)t * 64 + c] + v;
  }
}

// --------------------------------------------------------------- driver ----
extern "C" void kernel_launch(void* const* d_in, const int* in_sizes, int n_in,
                              void* d_out, int out_size, void* d_ws, size_t ws_size,
                              hipStream_t stream) {
  (void)in_sizes; (void)n_in; (void)out_size; (void)ws_size;
  const float* inp   = (const float*)d_in[0];
  const float* coord = (const float*)d_in[1];
  const float* cell  = (const float*)d_in[2];
  auto P = [&](int i) { return (const float*)d_in[i]; };
  // jax tree-flatten (alphabetical) order:
  // 3..12  attn0: b1,b2,bqkv,kln_b,kln_g,vln_b,vln_g,w1,w2,wqkv
  // 13..22 attn1 (same)
  // 23 enc_b, 24 enc_w
  // 25+6n  fc[n]: b1,b2,ln_b,ln_g,w1,w2
  // 61 freq_b, 62 freq_w

  char* ws = (char*)d_ws;
  const size_t MB = 1ull << 20;
  float*    feat = (float*)(ws + 0 * MB);     // 256x4096 f32      (4 MB)
  float*    enc  = (float*)(ws + 4 * MB);     // 64x4096 f32       (1 MB)
  float*    lc   = (float*)(ws + 5 * MB);     // 65536x2 f32
  int*      idx4 = (int*)  (ws + 6 * MB);     // 65536x4 i32
  float*    w4   = (float*)(ws + 7 * MB);     // 65536x4 f32
  float*    kv   = (float*)(ws + 8 * MB);     // 16x16x16 f32
  _Float16* w16  = (_Float16*)(ws + 9 * MB);  // weight staging    (<=1 MB)
  float*    X    = (float*)(ws + 16 * MB);    // 65536x256 f32     (64 MB)
  float*    Y    = (float*)(ws + 80 * MB);    // 65536x256 f32     (64 MB)
  _Float16* a16  = (_Float16*)(ws + 144 * MB);// 65536x1056 f16    (132 MB)
  _Float16* h16  = (_Float16*)(ws + 276 * MB);// 65536x768 f16     (96 MB)

  conv3x3_k<<<(64 * NPIX + 255) / 256, 256, 0, stream>>>(inp, P(24), P(23), enc, 3, 64);
  conv3x3_k<<<(256 * NPIX + 255) / 256, 256, 0, stream>>>(enc, P(62), P(61), feat, 64, 256);
  samp_k<<<HW_TOK / 256, 256, 0, stream>>>(coord, idx4, w4, lc);

  { // fc[0]: LN(1028) -> 256 (gelu) -> 256
    int fb = 25;
    ln_big<<<HW_TOK / 8, 256, 0, stream>>>(feat, idx4, w4, lc, cell, P(fb + 3), P(fb + 2), a16);
    wcvt_k<<<(1056 * 256 + 255) / 256, 256, 0, stream>>>(P(fb + 4), w16, 1028, 256, 1056, 256);
    gemm_wmma<<<dim3(512, 4), 256, 0, stream>>>(a16, w16, P(fb + 0), h16, nullptr, 1056, 256, 256, EPI_GELU16);
    wcvt_k<<<(256 * 256 + 255) / 256, 256, 0, stream>>>(P(fb + 5), w16, 256, 256, 256, 256);
    gemm_wmma<<<dim3(512, 4), 256, 0, stream>>>(h16, w16, P(fb + 1), X, nullptr, 256, 256, 256, EPI_F32);
  }

  for (int ai = 0; ai < 2; ++ai) { // two linear-attention blocks
    int ab = 3 + 10 * ai;
    cvt16_k<<<HW_TOK, 256, 0, stream>>>(X, a16, HW_TOK * 256);
    wcvt_k<<<(256 * 768 + 255) / 256, 256, 0, stream>>>(P(ab + 9), w16, 256, 768, 256, 768);
    gemm_wmma<<<dim3(512, 12), 256, 0, stream>>>(a16, w16, P(ab + 2), h16, nullptr, 256, 768, 768, EPI_F16);
    kv_zero_k<<<16, 256, 0, stream>>>(kv);
    kv_reduce<<<256, 256, 0, stream>>>(h16, P(ab + 4), P(ab + 3), P(ab + 6), P(ab + 5), kv, HW_TOK / 256);
    apply_o_k<<<HW_TOK, 256, 0, stream>>>(h16, kv, X, Y, a16);
    wcvt_k<<<(256 * 256 + 255) / 256, 256, 0, stream>>>(P(ab + 7), w16, 256, 256, 256, 256);
    gemm_wmma<<<dim3(512, 4), 256, 0, stream>>>(a16, w16, P(ab + 0), h16, nullptr, 256, 256, 256, EPI_GELU16);
    wcvt_k<<<(256 * 256 + 255) / 256, 256, 0, stream>>>(P(ab + 8), w16, 256, 256, 256, 256);
    gemm_wmma<<<dim3(512, 4), 256, 0, stream>>>(h16, w16, P(ab + 1), X, X, 256, 256, 256, EPI_F32);
  }

  for (int n = 1; n <= 4; ++n) { // fc[1..4]
    int fb = 25 + 6 * n;
    ln_small<<<HW_TOK / 8, 256, 0, stream>>>(X, lc, P(fb + 3), P(fb + 2), a16, n);
    wcvt_k<<<(288 * 256 + 255) / 256, 256, 0, stream>>>(P(fb + 4), w16, 258, 256, 288, 256);
    gemm_wmma<<<dim3(512, 4), 256, 0, stream>>>(a16, w16, P(fb + 0), h16, nullptr, 288, 256, 256, EPI_GELU16);
    wcvt_k<<<(256 * 256 + 255) / 256, 256, 0, stream>>>(P(fb + 5), w16, 256, 256, 256, 256);
    gemm_wmma<<<dim3(512, 4), 256, 0, stream>>>(h16, w16, P(fb + 1), X, nullptr, 256, 256, 256, EPI_F32);
  }

  { // fc[5]: -> 3 channels (N padded to 64)
    int fb = 25 + 30;
    ln_small<<<HW_TOK / 8, 256, 0, stream>>>(X, lc, P(fb + 3), P(fb + 2), a16, 5);
    wcvt_k<<<(288 * 256 + 255) / 256, 256, 0, stream>>>(P(fb + 4), w16, 258, 256, 288, 256);
    gemm_wmma<<<dim3(512, 4), 256, 0, stream>>>(a16, w16, P(fb + 0), h16, nullptr, 288, 256, 256, EPI_GELU16);
    wcvt_k<<<(256 * 64 + 255) / 256, 256, 0, stream>>>(P(fb + 5), w16, 256, 3, 256, 64);
    gemm_wmma<<<dim3(512, 1), 256, 0, stream>>>(h16, w16, P(fb + 1), Y, nullptr, 256, 64, 3, EPI_F32);
  }

  finalk_k<<<HW_TOK / 256, 256, 0, stream>>>(Y, inp, coord, (float*)d_out);
}
